// PaPs_22539988370118
// MI455X (gfx1250) — matrix operand order repaired and checked by reading, not compiled, not run on hardware
//
#include <hip/hip_runtime.h>
#include <hip/hip_bf16.h>
#include <math.h>

// ---------------- problem constants ----------------
#define BB 8
#define HH 256
#define WW 256
#define ENCC 64
#define SSS 16
#define KCLS 20
#define MPRO 256
#define STK 192
#define HWSZ (HH*WW)
#define EPSC 1e-5f
#define CANDCAP 65536
#define WPACK_ELEMS (2*2*18*32*16)   // [head][ot][kstep][lane][16] f16 fragments

// ---------------- WMMA vector types ----------------
typedef __attribute__((ext_vector_type(16))) _Float16 v16h;
typedef __attribute__((ext_vector_type(8)))  float    v8f;
union V16  { v16h v; _Float16 e[16]; };
union V16b { v16h v; uint4 q[2]; };          // 32-byte fragment as two b128 loads
union V8   { v8f  v; float    e[8];  };

__device__ __forceinline__ int refl(int i, int n){ return i < 0 ? -i : (i >= n ? 2*n - 2 - i : i); }
__device__ __forceinline__ float sigm(float x){ return 1.f/(1.f + __expf(-x)); }

struct Head1P { const float* w1; const float* b1; const float* g1; const float* be1; const float* m1; const float* v1; };
struct Head2P { const float* w2; const float* b2; const float* g2; const float* be2; const float* m2; const float* v2; };

// ======================================================================
// Kernel 0: pre-pack conv1 weights (both heads) into per-lane WMMA A-frag
// layout: idx = ((((h*2+ot)*18 + s)*32 + lane)*16 + j), f32 -> f16.
// 73 KB total -> stays L2-resident for all conv1 blocks.
// ======================================================================
__global__ void __launch_bounds__(256) k_packw1(Head1P ph, Head1P ps, _Float16* __restrict__ wpack)
{
  int g = blockIdx.x*256 + threadIdx.x;      // WPACK_ELEMS
  int j    = g & 15;
  int lane = (g >> 4) & 31;
  int s    = (g >> 9) % 18;
  int hot  = (g >> 9) / 18;
  int ot   = hot & 1;
  int h    = hot >> 1;
  const float* wp = h ? ps.w1 : ph.w1;       // (32,64,3,3) OIHW
  int ln16 = lane & 15, hi = lane >> 4;
  int o  = ot*16 + ln16;
  int kk = hi*16 + j;                        // contiguous k ordering (matches B fill)
  int tap = s >> 1, dy = tap/3, dx = tap%3;
  int cin = (s & 1)*32 + kk;
  wpack[g] = (_Float16)wp[((o*ENCC + cin)*3 + dy)*3 + dx];
}

// ======================================================================
// Kernel 1: conv3x3 64->32 (+bias+BN+ReLU), both heads fused, WMMA GEMM.
// Per block: one image row segment of 128 pixels (8 waves x 16 px).
// A = packed weights (b128 loads), B = im2col pixels from LDS (ds b128).
// ======================================================================
__global__ void __launch_bounds__(256) k_conv1(const float* __restrict__ xin,
                                               const _Float16* __restrict__ wpack,
                                               Head1P ph, Head1P ps,
                                               _Float16* __restrict__ y1)
{
  __shared__ _Float16 sIn[3*130*64];   // [row][col][cin], 49,920 bytes
  const int b  = blockIdx.z;
  const int y  = blockIdx.y;
  const int x0 = blockIdx.x * 128;
  const int tid = threadIdx.x;

  // stage 3 input rows (reflect pad), f32 -> f16
  for (int i = tid; i < 3*130*64; i += 256){
    int col = i % 130;
    int r   = (i / 130) % 3;
    int c   = i / 390;
    int yy  = refl(y + r - 1, HH);
    int xx  = refl(x0 + col - 1, WW);
    sIn[(r*130 + col)*64 + c] = (_Float16)xin[((b*ENCC + c)*HH + yy)*WW + xx];
  }
  __syncthreads();

  const int wave = tid >> 5, lane = tid & 31;
  const int ln16 = lane & 15, hi = lane >> 4;
  v8f acc[2][2] = {};                      // [head][out-tile]
  const Head1P P[2] = {ph, ps};

  for (int s = 0; s < 18; ++s){            // K = 18 steps of 32
    const int tap = s >> 1, dy = tap/3, dx = tap%3, cinb = (s & 1)*32;
    // prefetch next K-step's packed-weight line (global_prefetch_b8)
    if (s + 1 < 18)
      __builtin_prefetch(&wpack[(((size_t)(s + 1))*32 + lane)*16], 0, 1);
    // B fragment: column = pixel; contiguous 32B from LDS -> 2x ds_load_b128
    V16b bf;
    const uint4* bp = (const uint4*)(sIn + (dy*130 + (wave*16 + ln16 + dx))*64 + cinb + hi*16);
    bf.q[0] = bp[0]; bf.q[1] = bp[1];
    #pragma unroll
    for (int h = 0; h < 2; ++h){
      #pragma unroll
      for (int ot = 0; ot < 2; ++ot){
        // A fragment: pre-packed, contiguous 32B -> 2x global_load_b128
        V16b af;
        const uint4* ap = (const uint4*)(wpack + ((((h*2 + ot)*18 + s)*32 + lane)*16));
        af.q[0] = ap[0]; af.q[1] = ap[1];
        acc[h][ot] = __builtin_amdgcn_wmma_f32_16x16x32_f16(
            false, af.v, false, bf.v, (short)0, acc[h][ot], false, false);
      }
    }
  }

  const int px = x0 + wave*16 + ln16;
  #pragma unroll
  for (int h = 0; h < 2; ++h){
    #pragma unroll
    for (int ot = 0; ot < 2; ++ot){
      V8 a; a.v = acc[h][ot];
      #pragma unroll
      for (int v = 0; v < 8; ++v){
        int o = ot*16 + v + hi*8;          // ISA C/D layout
        float t = a.e[v] + P[h].b1[o];
        t = P[h].g1[o]*(t - P[h].m1[o])*rsqrtf(P[h].v1[o] + EPSC) + P[h].be1[o];
        t = fmaxf(t, 0.f);
        y1[(((b*2 + h)*32 + o)*HH + y)*WW + px] = (_Float16)t;
      }
    }
  }
}

// ======================================================================
// Kernel 2: conv3x3 32->1 + BN (+sigmoid for heatmap) per head.
// ======================================================================
__global__ void __launch_bounds__(256) k_conv2(const _Float16* __restrict__ y1,
                                               Head2P ph, Head2P ps,
                                               float* __restrict__ heat, float* __restrict__ sal)
{
  int g = blockIdx.x*256 + threadIdx.x;    // B*2*H*W
  int x = g % WW, y = (g / WW) % HH;
  int h = (g / HWSZ) & 1, b = g / (2*HWSZ);
  const Head2P P = h ? ps : ph;
  float sum = P.b2[0];
  const _Float16* base = y1 + (size_t)(b*2 + h)*32*HWSZ;
  for (int o = 0; o < 32; ++o){
    #pragma unroll
    for (int dy = 0; dy < 3; ++dy){
      int yy = refl(y + dy - 1, HH);
      #pragma unroll
      for (int dx = 0; dx < 3; ++dx){
        int xx = refl(x + dx - 1, WW);
        sum += (float)base[(o*HH + yy)*WW + xx] * P.w2[(o*3 + dy)*3 + dx];
      }
    }
  }
  float t = P.g2[0]*(sum - P.m2[0])*rsqrtf(P.v2[0] + EPSC) + P.be2[0];
  int off = b*HWSZ + y*WW + x;
  if (h == 0) heat[off] = sigm(t);
  else        sal[off]  = t;
}

// ======================================================================
// Stats / center mask / candidate compaction / top-k
// ======================================================================
__global__ void k_zero_stats(float* hsum, int* cc){ if (threadIdx.x == 0){ *hsum = 0.f; *cc = 0; } }

__global__ void __launch_bounds__(256) k_heatsum(const float* __restrict__ heat, float* hsum)
{
  __shared__ float sd[256];
  float s = 0.f;
  for (int i = blockIdx.x*256 + threadIdx.x; i < BB*HWSZ; i += gridDim.x*256) s += heat[i];
  sd[threadIdx.x] = s; __syncthreads();
  for (int st = 128; st > 0; st >>= 1){
    if (threadIdx.x < st) sd[threadIdx.x] += sd[threadIdx.x + st];
    __syncthreads();
  }
  if (threadIdx.x == 0) atomicAdd(hsum, sd[0]);
}

__global__ void __launch_bounds__(256) k_centermask(const float* __restrict__ heat,
                                                    float* __restrict__ cmask,
                                                    const float* hsum,
                                                    int* cand_count, int* cand_idx, float* cand_val)
{
  int g = blockIdx.x*256 + threadIdx.x;    // B*H*W
  int x = g % WW, y = (g / WW) % HH, b = g / HWSZ;
  float h = heat[g];
  float mx = -INFINITY;                    // pool pads with -inf (border ignored)
  for (int dy = -1; dy <= 1; ++dy){
    int yy = y + dy; if (yy < 0 || yy >= HH) continue;
    for (int dx = -1; dx <= 1; ++dx){
      int xx = x + dx; if (xx < 0 || xx >= WW) continue;
      mx = fmaxf(mx, heat[b*HWSZ + yy*WW + xx]);
    }
  }
  float mean = hsum[0] * (1.f/((float)BB*HWSZ));
  bool cm = (h == mx) && (h > mean);
  cmask[g] = cm ? 1.f : 0.f;
  if (cm){
    int slot = atomicAdd(cand_count, 1);
    if (slot < CANDCAP){ cand_idx[slot] = g; cand_val[slot] = h; }
  }
}

__global__ void __launch_bounds__(256) k_topk(const int* cand_count, const int* cand_idx, float* cand_val,
                                              int* tk_idx, float* tk_val,
                                              int* cbA, int* chA, int* cwA, int* validA,
                                              const float* __restrict__ heat, float* __restrict__ conf_out)
{
  __shared__ float sv[256]; __shared__ int sj[256];
  int n = cand_count[0]; if (n > CANDCAP) n = CANDCAP;
  int tid = threadIdx.x;
  for (int pass = 0; pass < MPRO; ++pass){
    float bv = -INFINITY; int bj = -1;
    for (int i = tid; i < n; i += 256){
      float v = cand_val[i];
      bool take = (v > bv) || (v == bv && bj >= 0 && v > -1e30f && cand_idx[i] < cand_idx[bj]);
      if (take){ bv = v; bj = i; }
    }
    sv[tid] = bv; sj[tid] = bj; __syncthreads();
    for (int st = 128; st > 0; st >>= 1){
      if (tid < st){
        float v2 = sv[tid+st]; int j2 = sj[tid+st];
        bool take = (v2 > sv[tid]) ||
                    (v2 == sv[tid] && j2 >= 0 && (sj[tid] < 0 ||
                     (v2 > -1e30f && cand_idx[j2] < cand_idx[sj[tid]])));
        if (take){ sv[tid] = v2; sj[tid] = j2; }
      }
      __syncthreads();
    }
    if (tid == 0){
      int j = sj[0];
      tk_val[pass] = sv[0];
      tk_idx[pass] = (j >= 0) ? cand_idx[j] : 0;
      if (j >= 0) cand_val[j] = -INFINITY;
    }
    __syncthreads();
  }
  if (tid < MPRO){
    float v = tk_val[tid];
    int flat = tk_idx[tid];
    int vld = (v > -1e30f) ? 1 : 0;
    int cb = flat / HWSZ, r = flat % HWSZ;
    cbA[tid] = cb; chA[tid] = r / WW; cwA[tid] = r % WW; validA[tid] = vld;
    conf_out[tid] = vld ? heat[flat] : 0.f;
  }
}

// ======================================================================
// Gather 192-dim feature stack for each proposal.
// ======================================================================
__global__ void __launch_bounds__(256) k_gather(const float* __restrict__ map0,
                                                const float* __restrict__ map1,
                                                const float* __restrict__ map2,
                                                const int* cbA, const int* chA, const int* cwA,
                                                float* __restrict__ stk)
{
  int g = blockIdx.x*256 + threadIdx.x;    // M*STK
  int m = g / STK, f = g % STK;
  int cb = cbA[m], ch = chA[m], cw = cwA[m];
  float v;
  if (f < 64)       v = map0[((cb*64 + f      )*(HH/4) + (ch>>2))*(WW/4) + (cw>>2)];
  else if (f < 128) v = map1[((cb*64 + (f-64) )*(HH/2) + (ch>>1))*(WW/2) + (cw>>1)];
  else              v = map2[((cb*64 + (f-128))*HH     +  ch    )*WW     +  cw    ];
  stk[m*STK + f] = v;
}

// ======================================================================
// Generic WMMA GEMM: out[256][N] = epi(A[256][Kd] @ Wt[Kd][N] + bias)
// mode 0: bias only; mode 1: bias + BN + ReLU. Kd % 32 == 0, N % 16 == 0.
// ======================================================================
__global__ void __launch_bounds__(256) k_gemm(const float* __restrict__ A, const float* __restrict__ Wt,
                                              const float* __restrict__ bias,
                                              const float* gg, const float* be, const float* mn, const float* vr,
                                              float* __restrict__ out, int Kd, int N, int mode)
{
  const int wave = threadIdx.x >> 5, lane = threadIdx.x & 31;
  const int ln16 = lane & 15, hi = lane >> 4;
  const int Ntiles = N >> 4;
  const int t = blockIdx.x*8 + wave;
  if (t >= 16*Ntiles) return;              // wave-uniform; EXEC stays full for WMMA
  const int mt = t / Ntiles, nt = t % Ntiles;
  const int row = mt*16 + ln16, col = nt*16 + ln16;
  v8f acc = {};
  for (int kb = 0; kb < Kd; kb += 32){
    V16 af, bf;
    #pragma unroll
    for (int j = 0; j < 16; ++j){
      int kk = (j & 7) + hi*8 + (j >> 3)*16;
      af.e[j] = (_Float16)A[row*Kd + kb + hi*16 + j];
      bf.e[j] = (_Float16)Wt[(kb + hi*16 + j)*N + col];
    }
    acc = __builtin_amdgcn_wmma_f32_16x16x32_f16(false, af.v, false, bf.v, (short)0, acc, false, false);
  }
  V8 a; a.v = acc;
  #pragma unroll
  for (int v = 0; v < 8; ++v){
    int r = mt*16 + v + hi*8;
    float tv = a.e[v] + bias[col];
    if (mode == 1){
      tv = gg[col]*(tv - mn[col])*rsqrtf(vr[col] + EPSC) + be[col];
      tv = fmaxf(tv, 0.f);
    }
    out[r*N + col] = tv;
  }
}

// small dense tail (N=2 softplus / N=20 plain)
__global__ void k_smalllin(const float* __restrict__ A, const float* __restrict__ Wt,
                           const float* __restrict__ bias, float* __restrict__ out,
                           int Kd, int N, int mode)
{
  int g = blockIdx.x*256 + threadIdx.x;
  if (g >= MPRO*N) return;
  int m = g / N, n = g % N;
  float s = bias[n];
  for (int k = 0; k < Kd; ++k) s += A[m*Kd + k]*Wt[k*N + n];
  if (mode == 2) s = fmaxf(s, 0.f) + log1pf(__expf(-fabsf(s)));   // softplus
  out[m*N + n] = s;
}

// ======================================================================
// Box prep: csize, start, clipped extents.
// ======================================================================
__global__ void k_prep(const float* __restrict__ size_out, const int* chA, const int* cwA,
                       int* csz, int* startA, int* hsA, int* heA, int* wsA, int* weA)
{
  int m = threadIdx.x; if (m >= MPRO) return;
  #pragma unroll
  for (int a = 0; a < 2; ++a){
    float s = size_out[m*2 + a];
    int c = (int)rintf(s); if (c < 1) c = 1;           // round-half-even
    int half = c >> 1, rem = c & 1;
    int pos = (a == 0) ? chA[m] : cwA[m];
    int st = pos - half, sp = pos + half + rem;
    csz[m*2 + a] = c; startA[m*2 + a] = st;
    int lim = (a == 0) ? HH : WW;
    int cs = st < 0 ? 0 : (st > lim ? lim : st);
    int ce = sp < 0 ? 0 : (sp > lim ? lim : sp);
    if (a == 0){ hsA[m] = cs; heA[m] = ce; } else { wsA[m] = cs; weA[m] = ce; }
  }
}

// ======================================================================
// Mask rasterization: bilinear 16x16 shape -> HxW, gated by box + saliency.
// 256 blocks per instance (HW = 65536 = 256*256).
// ======================================================================
__global__ void __launch_bounds__(256) k_masks(const float* __restrict__ shapes, const float* __restrict__ sal,
                                               const int* cbA, const int* validA,
                                               const int* csz, const int* startA,
                                               const int* hsA, const int* heA, const int* wsA, const int* weA,
                                               float* __restrict__ masks)
{
  __shared__ float sh[SSS*SSS];
  const int m = blockIdx.x >> 8;
  const int p = ((blockIdx.x & 255) << 8) + threadIdx.x;
  sh[threadIdx.x] = shapes[m*256 + threadIdx.x];
  __syncthreads();
  int y = p / WW, x = p % WW;
  float ny = (float)csz[m*2 + 0], nx = (float)csz[m*2 + 1];
  int sy = startA[m*2 + 0], sx = startA[m*2 + 1];
  float srcy = ((float)(y - sy) + 0.5f)*(16.f/ny) - 0.5f;
  srcy = fminf(fmaxf(srcy, 0.f), 15.f);
  int y0 = (int)floorf(srcy); int y1i = (y0 + 1 > 15) ? 15 : y0 + 1; float fy = srcy - (float)y0;
  float srcx = ((float)(x - sx) + 0.5f)*(16.f/nx) - 0.5f;
  srcx = fminf(fmaxf(srcx, 0.f), 15.f);
  int x0i = (int)floorf(srcx); int x1i = (x0i + 1 > 15) ? 15 : x0i + 1; float fx = srcx - (float)x0i;
  float g00 = sh[y0*16 + x0i], g01 = sh[y0*16 + x1i];
  float g10 = sh[y1i*16 + x0i], g11 = sh[y1i*16 + x1i];
  float val = (1.f - fy)*((1.f - fx)*g00 + fx*g01) + fy*((1.f - fx)*g10 + fx*g11);
  float mv = sigm(val)*sigm(sal[cbA[m]*HWSZ + p]);
  bool inb = (y >= hsA[m]) && (y < heA[m]) && (x >= wsA[m]) && (x < weA[m]) && validA[m];
  masks[(size_t)m*HWSZ + p] = inb ? mv : 0.f;
}

// ======================================================================
// Per-batch stable descending rank sort (M=256, O(M) per thread).
// ======================================================================
__global__ void __launch_bounds__(256) k_sort(const int* cbA, const int* validA, const float* conf,
                                              int* order)
{
  __shared__ float sc[MPRO];
  int b = blockIdx.x, i = threadIdx.x;
  float ci = (cbA[i] == b && validA[i]) ? conf[i] : -INFINITY;
  sc[i] = ci; __syncthreads();
  int rank = 0;
  for (int j = 0; j < MPRO; ++j){
    float cj = sc[j];
    if (cj > ci || (cj == ci && j < i)) rank++;
  }
  order[b*MPRO + rank] = i;
}

// ======================================================================
// Sequential NMS scan per batch: block-wide counts, then panoptic paint.
// ======================================================================
__global__ void __launch_bounds__(1024) k_nms(const float* __restrict__ masks, const float* __restrict__ sem,
                                              const int* order, const int* cbA, const int* validA,
                                              const float* conf,
                                              float* __restrict__ pano, float* __restrict__ semmap)
{
  __shared__ int ssz[1024]; __shared__ int snn[1024];
  int b = blockIdx.x, tid = threadIdx.x;
  float* pb = pano + (size_t)b*HWSZ;
  float* sm = semmap + (size_t)b*KCLS*HWSZ;
  for (int p = tid; p < HWSZ; p += 1024) pb[p] = 0.f;
  for (int i = tid; i < KCLS*HWSZ; i += 1024) sm[i] = 0.f;
  __syncthreads();
  for (int n = 0; n < MPRO; ++n){
    int i = order[b*MPRO + n];
    float c = (cbA[i] == b && validA[i]) ? conf[i] : -INFINITY;
    const float* mi = masks + (size_t)i*HWSZ;
    int lsz = 0, lnn = 0;
    for (int p = tid; p < HWSZ; p += 1024){
      if (mi[p] > 0.4f){ lsz++; if (pb[p] == 0.f) lnn++; }
    }
    ssz[tid] = lsz; snn[tid] = lnn; __syncthreads();
    for (int st = 512; st > 0; st >>= 1){
      if (tid < st){ ssz[tid] += ssz[tid + st]; snn[tid] += snn[tid + st]; }
      __syncthreads();
    }
    int szv = ssz[0], nnv = snn[0];
    bool acc = (c >= 0.2f) && (szv > 0) && ((float)nnv > 0.5f*(float)szv);
    __syncthreads();
    if (acc){
      for (int p = tid; p < HWSZ; p += 1024){
        if (mi[p] > 0.4f && pb[p] == 0.f){
          pb[p] = (float)(n + 1);
          for (int k = 0; k < KCLS; ++k) sm[(size_t)k*HWSZ + p] = sem[i*KCLS + k];
        }
      }
    }
    __syncthreads();
  }
}

// ======================================================================
// Host launcher
// ======================================================================
static inline size_t alignup(size_t x){ return (x + 255) & ~(size_t)255; }

extern "C" void kernel_launch(void* const* d_in, const int* in_sizes, int n_in,
                              void* d_out, int out_size, void* d_ws, size_t ws_size,
                              hipStream_t stream)
{
  (void)in_sizes; (void)n_in; (void)out_size; (void)ws_size;
  const float* xout = (const float*)d_in[0];
  const float* map0 = (const float*)d_in[1];
  const float* map1 = (const float*)d_in[2];
  const float* map2 = (const float*)d_in[3];
  #define PRM(i) ((const float*)d_in[4 + (i)])
  Head1P ph1 = {PRM(0), PRM(1), PRM(2), PRM(3), PRM(4), PRM(5)};
  Head2P ph2 = {PRM(6), PRM(7), PRM(8), PRM(9), PRM(10), PRM(11)};
  Head1P ps1 = {PRM(12), PRM(13), PRM(14), PRM(15), PRM(16), PRM(17)};
  Head2P ps2 = {PRM(18), PRM(19), PRM(20), PRM(21), PRM(22), PRM(23)};

  // ---- output layout (floats) ----
  float* dout = (float*)d_out;
  float* o_heat   = dout;                              // B*HW
  float* o_sal    = dout + 524288;                     // B*HW
  float* o_cmask  = dout + 1048576;                    // B*HW
  float* o_sem    = dout + 1572864;                    // M*K
  float* o_size   = dout + 1577984;                    // M*2
  float* o_conf   = dout + 1578496;                    // M
  float* o_masks  = dout + 1578752;                    // M*HW
  float* o_pano   = dout + 18355968;                   // B*HW
  float* o_semmap = dout + 18880256;                   // B*K*HW

  // ---- workspace layout ----
  char* wsb = (char*)d_ws;
  size_t off = 0;
  _Float16* y1    = (_Float16*)(wsb + off); off = alignup(off + (size_t)BB*2*32*HWSZ*2);
  _Float16* wpack = (_Float16*)(wsb + off); off = alignup(off + (size_t)WPACK_ELEMS*2);
  float* hsum      = (float*)(wsb + off); off = alignup(off + 4);
  int*   cand_cnt  = (int*)  (wsb + off); off = alignup(off + 4);
  int*   cand_idx  = (int*)  (wsb + off); off = alignup(off + (size_t)CANDCAP*4);
  float* cand_val  = (float*)(wsb + off); off = alignup(off + (size_t)CANDCAP*4);
  int*   tk_idx    = (int*)  (wsb + off); off = alignup(off + MPRO*4);
  float* tk_val    = (float*)(wsb + off); off = alignup(off + MPRO*4);
  int*   cbA       = (int*)  (wsb + off); off = alignup(off + MPRO*4);
  int*   chA       = (int*)  (wsb + off); off = alignup(off + MPRO*4);
  int*   cwA       = (int*)  (wsb + off); off = alignup(off + MPRO*4);
  int*   validA    = (int*)  (wsb + off); off = alignup(off + MPRO*4);
  float* stk       = (float*)(wsb + off); off = alignup(off + (size_t)MPRO*STK*4);
  float* y96a      = (float*)(wsb + off); off = alignup(off + (size_t)MPRO*96*4);
  float* shapes    = (float*)(wsb + off); off = alignup(off + (size_t)MPRO*256*4);
  float* y96b      = (float*)(wsb + off); off = alignup(off + (size_t)MPRO*96*4);
  float* y48b      = (float*)(wsb + off); off = alignup(off + (size_t)MPRO*48*4);
  float* y96c      = (float*)(wsb + off); off = alignup(off + (size_t)MPRO*96*4);
  float* y48c      = (float*)(wsb + off); off = alignup(off + (size_t)MPRO*48*4);
  int*   cszA      = (int*)  (wsb + off); off = alignup(off + MPRO*2*4);
  int*   startA    = (int*)  (wsb + off); off = alignup(off + MPRO*2*4);
  int*   hsA       = (int*)  (wsb + off); off = alignup(off + MPRO*4);
  int*   heA       = (int*)  (wsb + off); off = alignup(off + MPRO*4);
  int*   wsA       = (int*)  (wsb + off); off = alignup(off + MPRO*4);
  int*   weA       = (int*)  (wsb + off); off = alignup(off + MPRO*4);
  int*   order     = (int*)  (wsb + off); off = alignup(off + (size_t)BB*MPRO*4);

  // ---- pipeline ----
  k_packw1<<<WPACK_ELEMS/256, 256, 0, stream>>>(ph1, ps1, wpack);
  k_conv1<<<dim3(WW/128, HH, BB), 256, 0, stream>>>(xout, wpack, ph1, ps1, y1);
  k_conv2<<<(BB*2*HWSZ)/256, 256, 0, stream>>>(y1, ph2, ps2, o_heat, o_sal);

  k_zero_stats<<<1, 32, 0, stream>>>(hsum, cand_cnt);
  k_heatsum<<<512, 256, 0, stream>>>(o_heat, hsum);
  k_centermask<<<(BB*HWSZ)/256, 256, 0, stream>>>(o_heat, o_cmask, hsum, cand_cnt, cand_idx, cand_val);
  k_topk<<<1, 256, 0, stream>>>(cand_cnt, cand_idx, cand_val, tk_idx, tk_val,
                                cbA, chA, cwA, validA, o_heat, o_conf);
  k_gather<<<(MPRO*STK)/256, 256, 0, stream>>>(map0, map1, map2, cbA, chA, cwA, stk);

  // shape head: 192 -> 96 (BN+ReLU) -> 256
  k_gemm<<<12, 256, 0, stream>>>(stk, PRM(24), PRM(25), PRM(26), PRM(27), PRM(28), PRM(29),
                                 y96a, STK, 96, 1);
  k_gemm<<<32, 256, 0, stream>>>(y96a, PRM(30), PRM(31), nullptr, nullptr, nullptr, nullptr,
                                 shapes, 96, 256, 0);
  // size head: 192 -> 96 -> 48 (BN+ReLU each) -> 2 (softplus)
  k_gemm<<<12, 256, 0, stream>>>(stk, PRM(32), PRM(33), PRM(34), PRM(35), PRM(36), PRM(37),
                                 y96b, STK, 96, 1);
  k_gemm<<<6, 256, 0, stream>>>(y96b, PRM(38), PRM(39), PRM(40), PRM(41), PRM(42), PRM(43),
                                y48b, 96, 48, 1);
  k_smalllin<<<2, 256, 0, stream>>>(y48b, PRM(44), PRM(45), o_size, 48, 2, 2);
  // class head: 192 -> 96 (BN+ReLU) -> 48 -> 20
  k_gemm<<<12, 256, 0, stream>>>(stk, PRM(46), PRM(47), PRM(48), PRM(49), PRM(50), PRM(51),
                                 y96c, STK, 96, 1);
  k_gemm<<<6, 256, 0, stream>>>(y96c, PRM(52), PRM(53), nullptr, nullptr, nullptr, nullptr,
                                y48c, 96, 48, 0);
  k_smalllin<<<20, 256, 0, stream>>>(y48c, PRM(54), PRM(55), o_sem, 48, KCLS, 0);

  k_prep<<<1, 256, 0, stream>>>(o_size, chA, cwA, cszA, startA, hsA, heA, wsA, weA);
  k_masks<<<MPRO*256, 256, 0, stream>>>(shapes, o_sal, cbA, validA, cszA, startA,
                                        hsA, heA, wsA, weA, o_masks);
  k_sort<<<BB, 256, 0, stream>>>(cbA, validA, o_conf, order);
  k_nms<<<BB, 1024, 0, stream>>>(o_masks, o_sem, order, cbA, validA, o_conf, o_pano, o_semmap);
  #undef PRM
}
